// QuantumGeometricAttention_56109452755033
// MI455X (gfx1250) — matrix operand order repaired and checked by reading, not compiled, not run on hardware
//
#include <hip/hip_runtime.h>

typedef _Float16 half_t;
typedef __attribute__((ext_vector_type(8)))  _Float16 v8h_t;
typedef __attribute__((ext_vector_type(16))) _Float16 v16h;
typedef __attribute__((ext_vector_type(8)))  float    v8f;
typedef __attribute__((ext_vector_type(4)))  int      v4i;

#define HIDDEN   512
#define HEADS    8
#define HEAD_DIM 64
#define MAN      256
#define SEQ      2048
#define BATCH    2
#define TOKENS   4096
#define QK_SCALE 0.125f

// ---------------------------------------------------------------------------
// Async global->LDS staging (CDNA5 GLOBAL_LOAD_ASYNC_TO_LDS_B128), guarded.
// Builtin signature (from hipcc diagnostic): (global int4*, lds int4*, imm, imm)
// ---------------------------------------------------------------------------
#if defined(__gfx1250__) && __has_builtin(__builtin_amdgcn_global_load_async_to_lds_b128) && __has_builtin(__builtin_amdgcn_s_wait_asynccnt)
#define HAVE_ASYNC_LDS 1
typedef __attribute__((address_space(1))) v4i g_v4i;
typedef __attribute__((address_space(3))) v4i l_v4i;
__device__ __forceinline__ void cp16_async(void* lds_dst, const void* g_src) {
  __builtin_amdgcn_global_load_async_to_lds_b128((g_v4i*)g_src, (l_v4i*)lds_dst,
                                                 0, 0);
}
__device__ __forceinline__ void cp_wait() { __builtin_amdgcn_s_wait_asynccnt(0); }
#else
#define HAVE_ASYNC_LDS 0
__device__ __forceinline__ void cp16_async(void* lds_dst, const void* g_src) {
  *(uint4*)lds_dst = *(const uint4*)g_src;
}
__device__ __forceinline__ void cp_wait() {}
#endif

// ---------------------------------------------------------------------------
// WMMA fragment helpers (layouts per CDNA5 ISA 7.12.2, wave32)
// ---------------------------------------------------------------------------
__device__ __forceinline__ v8f wmma_f32_16x16x32(v16h a, v16h b, v8f c) {
  // (neg_a, A, neg_b, B, c_mod, C, reuse_a, reuse_b)
  return __builtin_amdgcn_wmma_f32_16x16x32_f16(false, a, false, b, (short)0, c,
                                                false, false);
}

// A matrix 16x32 f16: lanes 0-15 -> rows M=0..15 with K {0..7,16..23};
// lanes 16-31 -> same rows with K {8..15,24..31}.
__device__ __forceinline__ v16h frag_a16x32(const half_t* base, int ld, int mrow,
                                            int kbase) {
  int lane = threadIdx.x & 31;
  int row  = mrow + (lane & 15);
  int koff = (lane >> 4) * 8;
  const half_t* p0 = base + (size_t)row * ld + kbase + koff;
  v8h_t lo = *(const v8h_t*)p0;
  v8h_t hi = *(const v8h_t*)(p0 + 16);
  return __builtin_shufflevector(lo, hi, 0, 1, 2, 3, 4, 5, 6, 7,
                                 8, 9, 10, 11, 12, 13, 14, 15);
}

// B matrix 32x16 f16: lane l holds column N=l%16 with K {0..7,16..23} (lanes 0-15)
// or K {8..15,24..31} (lanes 16-31), matching the A layout.
__device__ __forceinline__ v16h frag_b32x16(const half_t* base, int ld, int kbase,
                                            int ncol) {
  int lane = threadIdx.x & 31;
  int col = ncol + (lane & 15);
  int koff = kbase + (lane >> 4) * 8;
  v16h r;
#pragma unroll
  for (int e = 0; e < 8; ++e) {
    r[e] = base[(size_t)(koff + e) * ld + col];
    r[8 + e] = base[(size_t)(koff + 16 + e) * ld + col];
  }
  return r;
}

// ---------------------------------------------------------------------------
// Prologue 1: fold complex head-split + manifold projection into one f16
// weight: WIN[512][12288], n = mat*4096 + h*512 + c*256 + m  (c: 0=re 1=im)
// ---------------------------------------------------------------------------
__global__ __launch_bounds__(256) void k_prep_win(
    const float* __restrict__ Wq_re, const float* __restrict__ Wq_im,
    const float* __restrict__ Wk_re, const float* __restrict__ Wk_im,
    const float* __restrict__ Wv_re, const float* __restrict__ Wv_im,
    const float* __restrict__ Wm_re, const float* __restrict__ Wm_im,
    half_t* __restrict__ win) {
  long o = (long)blockIdx.x * 256 + threadIdx.x;  // < 512*12288
  int d = (int)(o / 12288);
  int n = (int)(o % 12288);
  int mat = n >> 12;           // /4096
  int rem = n & 4095;
  int h = rem >> 9;
  int rem2 = rem & 511;
  int c = rem2 >> 8;
  int m = rem2 & 255;
  const float* Wre = (mat == 0) ? Wq_re : (mat == 1 ? Wk_re : Wv_re);
  const float* Wim = (mat == 0) ? Wq_im : (mat == 1 ? Wk_im : Wv_im);
  const float* ar = Wre + (size_t)d * HIDDEN + h * HEAD_DIM;
  const float* ai = Wim + (size_t)d * HIDDEN + h * HEAD_DIM;
  float acc = 0.f;
  for (int e = 0; e < HEAD_DIM; ++e) {
    float br = Wm_re[e * MAN + m], bi = Wm_im[e * MAN + m];
    acc += (c == 0) ? (ar[e] * br - ai[e] * bi) : (ar[e] * bi + ai[e] * br);
  }
  win[o] = (half_t)acc;
}

// ---------------------------------------------------------------------------
// Prologue 2: fold Wmi @ Wo into W2[4096][1024]:
//  k = h*512 + c*256 + m ; n = c2*512 + j (c2: 0 -> out_re, 1 -> out_im)
//  out_re uses [Wmo_re ; -Wmo_im], out_im uses [Wmo_im ; Wmo_re]
// ---------------------------------------------------------------------------
__global__ __launch_bounds__(256) void k_prep_w2(
    const float* __restrict__ Wmi_re, const float* __restrict__ Wmi_im,
    const float* __restrict__ Wo_re, const float* __restrict__ Wo_im,
    half_t* __restrict__ w2) {
  long o = (long)blockIdx.x * 256 + threadIdx.x;  // < 4096*1024
  int k = (int)(o >> 10);
  int n = (int)(o & 1023);
  int h = k >> 9;
  int r = k & 511;
  int c = r >> 8;
  int m = r & 255;
  int c2 = n >> 9;
  int j = n & 511;
  float re = 0.f, im = 0.f;
  for (int d2 = 0; d2 < HEAD_DIM; ++d2) {
    float mr = Wmi_re[m * HEAD_DIM + d2], mi = Wmi_im[m * HEAD_DIM + d2];
    float wr = Wo_re[(size_t)(h * HEAD_DIM + d2) * HIDDEN + j];
    float wi = Wo_im[(size_t)(h * HEAD_DIM + d2) * HIDDEN + j];
    re += mr * wr - mi * wi;
    im += mr * wi + mi * wr;
  }
  float val = (c2 == 0) ? ((c == 0) ? re : -im) : ((c == 0) ? im : re);
  w2[o] = (half_t)val;
}

// ---------------------------------------------------------------------------
// Row L2 normalization: xn = x / ||x||_2 (f32 -> f16), one token per block.
// ---------------------------------------------------------------------------
__global__ __launch_bounds__(256) void k_normalize(const float* __restrict__ x,
                                                   half_t* __restrict__ xn) {
  __shared__ float red[256];
  int t = blockIdx.x;
  int tid = threadIdx.x;
  const float* row = x + (size_t)t * HIDDEN;
  float v0 = row[tid], v1 = row[tid + 256];
  red[tid] = v0 * v0 + v1 * v1;
  __syncthreads();
  for (int off = 128; off > 0; off >>= 1) {
    if (tid < off) red[tid] += red[tid + off];
    __syncthreads();
  }
  float rn = rsqrtf(red[0]);
  xn[(size_t)t * HIDDEN + tid] = (half_t)(v0 * rn);
  xn[(size_t)t * HIDDEN + tid + 256] = (half_t)(v1 * rn);
}

// ---------------------------------------------------------------------------
// Shared 128x128 WMMA GEMM core (8 waves as 2(M)x4(N), wave tile 64x32, BK=64)
// ---------------------------------------------------------------------------
#define BK 64
__device__ __forceinline__ void gemm_core_128x128(
    const half_t* __restrict__ A, const half_t* __restrict__ B, int lda, int ldb,
    int Kdim, int mb, int nb, half_t* As, half_t* Bs, v8f acc[4][2]) {
  int tid = threadIdx.x;
  int wid = tid >> 5;
  int wm = (wid & 1) * 64;
  int wn = (wid >> 1) * 32;
  for (int kk = 0; kk < Kdim; kk += BK) {
#pragma unroll
    for (int i = 0; i < 4; ++i) {  // A tile: 128x64 halves, 16B per copy
      int idx = (tid * 4 + i) * 8;
      int r = idx >> 6, cc = idx & 63;
      cp16_async(&As[idx], &A[(size_t)(mb * 128 + r) * lda + kk + cc]);
    }
#pragma unroll
    for (int i = 0; i < 4; ++i) {  // B tile: 64x128 halves
      int idx = (tid * 4 + i) * 8;
      int r = idx >> 7, cc = idx & 127;
      cp16_async(&Bs[idx], &B[(size_t)(kk + r) * ldb + nb * 128 + cc]);
    }
    cp_wait();
    __syncthreads();
#pragma unroll
    for (int ks = 0; ks < BK; ks += 32) {
      v16h af[4], bf[2];
#pragma unroll
      for (int i = 0; i < 4; ++i) af[i] = frag_a16x32(As, BK, wm + i * 16, ks);
#pragma unroll
      for (int j = 0; j < 2; ++j) bf[j] = frag_b32x16(Bs, 128, ks, wn + j * 16);
#pragma unroll
      for (int i = 0; i < 4; ++i)
#pragma unroll
        for (int j = 0; j < 2; ++j)
          acc[i][j] = wmma_f32_16x16x32(af[i], bf[j], acc[i][j]);
    }
    __syncthreads();
  }
}

// GEMM1: XN[4096,512] x WIN[512,12288] -> QKVM f16 [mat][b][h][s][512]
__global__ __launch_bounds__(256) void k_gemm_qkv(const half_t* __restrict__ A,
                                                  const half_t* __restrict__ B,
                                                  half_t* __restrict__ qkvm) {
  __shared__ half_t As[128 * BK];
  __shared__ half_t Bs[BK * 128];
  int nb = blockIdx.x, mb = blockIdx.y;
  v8f acc[4][2] = {};
  gemm_core_128x128(A, B, HIDDEN, 12288, HIDDEN, mb, nb, As, Bs, acc);
  int tid = threadIdx.x, lane = tid & 31, wid = tid >> 5;
  int wm = (wid & 1) * 64, wn = (wid >> 1) * 32;
  int crow = lane >> 4, ccol = lane & 15;
#pragma unroll
  for (int i = 0; i < 4; ++i)
#pragma unroll
    for (int j = 0; j < 2; ++j)
#pragma unroll
      for (int v = 0; v < 8; ++v) {
        int token = mb * 128 + wm + i * 16 + v + crow * 8;
        int n = nb * 128 + wn + j * 16 + ccol;
        int mat = n >> 12, rem = n & 4095;
        int h = rem >> 9, p = rem & 511;
        int b = token >> 11, s = token & 2047;
        size_t off = ((((size_t)mat * BATCH + b) * HEADS + h) * SEQ + s) * 512 + p;
        qkvm[off] = (half_t)acc[i][j][v];
      }
}

// GEMM2: OM[4096,4096] x W2[4096,1024] -> f32 out, re/im interleaved last axis
__global__ __launch_bounds__(256) void k_gemm_out(const half_t* __restrict__ A,
                                                  const half_t* __restrict__ B,
                                                  float* __restrict__ out) {
  __shared__ half_t As[128 * BK];
  __shared__ half_t Bs[BK * 128];
  int nb = blockIdx.x, mb = blockIdx.y;
  v8f acc[4][2] = {};
  gemm_core_128x128(A, B, 4096, 1024, 4096, mb, nb, As, Bs, acc);
  int tid = threadIdx.x, lane = tid & 31, wid = tid >> 5;
  int wm = (wid & 1) * 64, wn = (wid >> 1) * 32;
  int crow = lane >> 4, ccol = lane & 15;
#pragma unroll
  for (int i = 0; i < 4; ++i)
#pragma unroll
    for (int j = 0; j < 2; ++j)
#pragma unroll
      for (int v = 0; v < 8; ++v) {
        int token = mb * 128 + wm + i * 16 + v + crow * 8;
        int n = nb * 128 + wn + j * 16 + ccol;
        int c2 = n >> 9, jj = n & 511;
        out[((size_t)token * HIDDEN + jj) * 2 + c2] = acc[i][j][v];
      }
}

// ---------------------------------------------------------------------------
// Attention: one workgroup = 16 query rows of one (b,h).
//  - Q^T tile in LDS (f16), full 16x2048 score block in LDS (f32),
//  - exact softmax (scale 1/8) in place, then P @ [V_re||V_im].
// Dynamic LDS: 16KB qt + 128KB S + 1KB red + 64B rowstat = 148544 B.
// ---------------------------------------------------------------------------
#define ATT_SMEM_BYTES (16 * 512 * 2 + 16 * SEQ * 4 + 256 * 4 + 16 * 4)
__global__ __launch_bounds__(256) void k_attn(const half_t* __restrict__ qkvm,
                                              half_t* __restrict__ om) {
  extern __shared__ char smem[];
  half_t* qt = (half_t*)smem;                        // [512][16]  Q^T
  float* S = (float*)(smem + 16 * 512 * 2);          // [16][2048] scores/probs
  float* red = (float*)(smem + 16 * 512 * 2 + 16 * SEQ * 4);  // [16][16]
  float* rowstat = red + 256;                        // [16]

  int qb = blockIdx.x;            // 0..127 (query block)
  int bh = blockIdx.y;            // 0..15
  int b = bh >> 3, h = bh & 7;
  size_t headsz = (size_t)SEQ * 512;
  const half_t* QM = qkvm + (((size_t)0 * BATCH + b) * HEADS + h) * headsz;
  const half_t* KM = qkvm + (((size_t)1 * BATCH + b) * HEADS + h) * headsz;
  const half_t* VM = qkvm + (((size_t)2 * BATCH + b) * HEADS + h) * headsz;
  int tid = threadIdx.x, lane = tid & 31, wid = tid >> 5;
  int qbase = qb * 16;

  // Stage Q^T (16 rows x 512) into LDS.
  for (int i = tid * 8; i < 16 * 512; i += 256 * 8) {
    int q = i >> 9, k = i & 511;
    uint4 d = *(const uint4*)&QM[(size_t)(qbase + q) * 512 + k];
    const half_t* hp = (const half_t*)&d;
#pragma unroll
    for (int e = 0; e < 8; ++e) qt[(k + e) * 16 + q] = hp[e];
  }
  __syncthreads();

  // Scores: wave 'wid' computes S^T tiles for kv range [wid*256, wid*256+256).
  int kv0 = wid * 256;
  for (int g = 0; g < 4; ++g) {  // groups of 4 kv tiles (64 kv rows)
    v8f acc[4] = {};
    int kvg = kv0 + g * 64;
    for (int ks = 0; ks < 512; ks += 32) {
      v16h bf = frag_b32x16(qt, 16, ks, 0);  // B = Q^T slice
#pragma unroll
      for (int t2 = 0; t2 < 4; ++t2) {       // A = K rows (global, L2-resident)
        v16h af = frag_a16x32(KM, 512, kvg + t2 * 16, ks);
        acc[t2] = wmma_f32_16x16x32(af, bf, acc[t2]);
      }
    }
    int crow = lane >> 4, ccol = lane & 15;
#pragma unroll
    for (int t2 = 0; t2 < 4; ++t2)
#pragma unroll
      for (int v = 0; v < 8; ++v)
        S[ccol * SEQ + kvg + t2 * 16 + v + crow * 8] = acc[t2][v];
  }
  __syncthreads();

  // Exact softmax over 2048: 16 rows x 16 segments of 128.
  int r = tid & 15, seg = tid >> 4;
  float* srow = S + r * SEQ + seg * 128;
  float mx = -3.0e38f;
  for (int i2 = 0; i2 < 128; ++i2) mx = fmaxf(mx, srow[i2]);
  red[r * 16 + seg] = mx;
  __syncthreads();
  if (tid < 16) {
    float m2 = red[tid * 16];
    for (int i2 = 1; i2 < 16; ++i2) m2 = fmaxf(m2, red[tid * 16 + i2]);
    rowstat[tid] = m2;
  }
  __syncthreads();
  float rm = rowstat[r];
  float sum = 0.f;
  for (int i2 = 0; i2 < 128; ++i2) {
    float e = __expf((srow[i2] - rm) * QK_SCALE);
    srow[i2] = e;  // in-place exp(scale*(s-m)); normalize in epilogue
    sum += e;
  }
  __syncthreads();
  red[r * 16 + seg] = sum;
  __syncthreads();
  if (tid < 16) {
    float s2 = 0.f;
    for (int i2 = 0; i2 < 16; ++i2) s2 += red[tid * 16 + i2];
    rowstat[tid] = 1.f / s2;
  }
  __syncthreads();

  // P @ V: wave 'wid' owns output cols [wid*64, wid*64+64).
  v8f oacc[4] = {};
  for (int kv = 0; kv < SEQ; kv += 32) {
    // A fragment from P (f32 in LDS -> f16), ISA A-layout K pattern.
    int prow = lane & 15;
    int koff = (lane >> 4) * 8;
    const float* p0 = S + prow * SEQ + kv + koff;
    half_t atmp[16];
#pragma unroll
    for (int e = 0; e < 8; ++e) {
      atmp[e] = (half_t)p0[e];
      atmp[8 + e] = (half_t)p0[16 + e];
    }
    v16h af = *(v16h*)atmp;
#pragma unroll
    for (int j = 0; j < 4; ++j) {
      v16h bf = frag_b32x16(VM, 512, kv, wid * 64 + j * 16);
      oacc[j] = wmma_f32_16x16x32(af, bf, oacc[j]);
    }
  }
  int crow = lane >> 4, ccol = lane & 15;
#pragma unroll
  for (int j = 0; j < 4; ++j)
#pragma unroll
    for (int v = 0; v < 8; ++v) {
      int mrow = v + crow * 8;
      float val = oacc[j][v] * rowstat[mrow];
      int token = b * SEQ + qbase + mrow;
      int p = wid * 64 + j * 16 + ccol;
      om[(size_t)token * 4096 + h * 512 + p] = (half_t)val;
    }
}

// ---------------------------------------------------------------------------
// Launch. Workspace layout (assumes ws_size >= 152 MiB):
//  [0,4M)  XN f16 | [4M,16M) WIN f16 | [16M,24M) W2 f16
//  [24M,120M) QKVM f16 | [120M,152M) OM f16
// ---------------------------------------------------------------------------
extern "C" void kernel_launch(void* const* d_in, const int* in_sizes, int n_in,
                              void* d_out, int out_size, void* d_ws,
                              size_t ws_size, hipStream_t stream) {
  (void)in_sizes; (void)n_in; (void)out_size; (void)ws_size;
  const float* x = (const float*)d_in[0];
  const float* Wq_re = (const float*)d_in[1];
  const float* Wq_im = (const float*)d_in[2];
  const float* Wk_re = (const float*)d_in[3];
  const float* Wk_im = (const float*)d_in[4];
  const float* Wv_re = (const float*)d_in[5];
  const float* Wv_im = (const float*)d_in[6];
  const float* Wm_re = (const float*)d_in[7];
  const float* Wm_im = (const float*)d_in[8];
  const float* Wmi_re = (const float*)d_in[9];
  const float* Wmi_im = (const float*)d_in[10];
  const float* Wo_re = (const float*)d_in[11];
  const float* Wo_im = (const float*)d_in[12];

  char* ws = (char*)d_ws;
  half_t* XN = (half_t*)(ws + 0);
  half_t* WIN = (half_t*)(ws + (4ll << 20));
  half_t* W2 = (half_t*)(ws + (16ll << 20));
  half_t* QKVM = (half_t*)(ws + (24ll << 20));
  half_t* OM = (half_t*)(ws + (120ll << 20));
  float* out = (float*)d_out;

  k_prep_win<<<24576, 256, 0, stream>>>(Wq_re, Wq_im, Wk_re, Wk_im, Wv_re,
                                        Wv_im, Wm_re, Wm_im, WIN);
  k_prep_w2<<<16384, 256, 0, stream>>>(Wmi_re, Wmi_im, Wo_re, Wo_im, W2);
  k_normalize<<<4096, 256, 0, stream>>>(x, XN);
  k_gemm_qkv<<<dim3(96, 32), 256, 0, stream>>>(XN, WIN, QKVM);
  k_attn<<<dim3(128, 16), 256, ATT_SMEM_BYTES, stream>>>(QKVM, OM);
  k_gemm_out<<<dim3(8, 32), 256, 0, stream>>>(OM, W2, out);
}
